// NONA_5093831213645
// MI455X (gfx1250) — matrix operand order repaired and checked by reading, not compiled, not run on hardware
//
#include <hip/hip_runtime.h>
#include <hip/hip_bf16.h>

// MI455X / gfx1250, wave32. Fused -cdist -> softmax -> @ y_n.
// GEMM1: split-bf16 (hi+lo) WMMA, B tiles async-staged to LDS (double buffer,
// shared by all 8 waves of the block). GEMM2: bf16 WMMA vs [Y|1]^T.
// Deterministic split-K partials (no atomics), reduce kernel normalizes.

typedef __attribute__((ext_vector_type(16))) __bf16 v16bf;
typedef __attribute__((ext_vector_type(8)))  float  v8f;
typedef __attribute__((ext_vector_type(4)))  int    v4i;

#if defined(__has_builtin)
#if __has_builtin(__builtin_amdgcn_global_load_async_to_lds_b128)
#define HAVE_ASYNC_LDS 1
#endif
#endif
#ifndef HAVE_ASYNC_LDS
#define HAVE_ASYNC_LDS 0
#endif

#define N_      8192
#define D_      1024
#define C_      100
#define CP_     112     // 100 classes + ones column (idx 100) + 11 zero pad
#define NSPLIT_ 8
#define BN_     64      // column tile per pass
#define KB_     64      // K chunk staged to LDS
#define KBP_    72      // padded LDS pitch (144B: 16B-aligned, conflict-free)
#define WAVES_  8

union Frag { v16bf bf; v4i q[2]; };

__device__ __forceinline__ v4i ld16(const __bf16* p) { return *(const v4i*)p; }

// Cooperative stage of one (hi,lo) B chunk: 64 cols x 64 K x 2 matrices,
// 1024 x 16B transfers across 256 threads (4 per thread).
__device__ __forceinline__ void stage_chunk(const __bf16* __restrict__ bh,
                                            const __bf16* __restrict__ bl,
                                            __bf16* bsb, int cb, int k0, int tid) {
#pragma unroll
  for (int i = 0; i < 4; ++i) {
    const int cid = tid + (i << 8);        // 0..1023
    const int m   = cid >> 9;              // 0 = hi, 1 = lo
    const int rem = cid & 511;
    const int col = rem >> 3;
    const int k8  = (rem & 7) << 3;
    const __bf16* g = (m ? bl : bh) + (size_t)(cb + col) * D_ + k0 + k8;
    __bf16* l = bsb + ((size_t)(m * BN_ + col) * KBP_ + k8);
#if HAVE_ASYNC_LDS
    __builtin_amdgcn_global_load_async_to_lds_b128(
        (__attribute__((address_space(1))) v4i*)(uintptr_t)g,
        (__attribute__((address_space(3))) v4i*)l, 0, 0);
#else
    *(v4i*)l = *(const v4i*)g;
#endif
  }
}

// ---- prep: f32 -> (hi,lo) bf16 split + row squared norms ------------------
__global__ void prep_convert(const float* __restrict__ x,
                             __bf16* __restrict__ hi, __bf16* __restrict__ lo,
                             float* __restrict__ nrm) {
  const int wave = threadIdx.x >> 5, lane = threadIdx.x & 31;
  const int row  = blockIdx.x * 8 + wave;
  const float* xr = x + (size_t)row * D_;
  float acc = 0.f;
  for (int k = lane; k < D_; k += 32) {
    float v = xr[k];
    __bf16 h = (__bf16)v;
    hi[(size_t)row * D_ + k] = h;
    lo[(size_t)row * D_ + k] = (__bf16)(v - (float)h);
    acc += v * v;
  }
  for (int off = 16; off > 0; off >>= 1) acc += __shfl_xor(acc, off, 32);
  if (lane == 0) nrm[row] = acc;
}

// ---- prep: Y^T padded with ones column, bf16 (exact: one-hot) -------------
__global__ void prep_ypadT(const float* __restrict__ y, __bf16* __restrict__ yt) {
  const int idx = blockIdx.x * blockDim.x + threadIdx.x;   // [0, CP_*N_)
  const int c = idx / N_, j = idx % N_;
  float v = (c < C_) ? y[(size_t)j * C_ + c] : (c == C_ ? 1.f : 0.f);
  yt[idx] = (__bf16)v;
}

// ---- main fused kernel -----------------------------------------------------
__global__ void __launch_bounds__(256) dist_softmax_wmma(
    const __bf16* __restrict__ xa_hi, const __bf16* __restrict__ xa_lo,
    const __bf16* __restrict__ xb_hi, const __bf16* __restrict__ xb_lo,
    const float*  __restrict__ x2a,   const float*  __restrict__ x2b,
    const __bf16* __restrict__ yT,    float* __restrict__ part) {
  __shared__ __align__(16) __bf16 Bs[2][2][BN_][KBP_]; // double-buffered B (hi,lo)
  __shared__ __align__(16) __bf16 Pb[WAVES_][16][72];  // per-wave P tile
  const int tid   = threadIdx.x;
  const int wave  = tid >> 5, lane = tid & 31;
  // Block = 8 adjacent row strips sharing ONE column split -> B/Y tiles are
  // identical across all 8 waves (LDS/L1 reuse on the dominant streams).
  const int split = blockIdx.x >> 6;
  const int r0    = (((blockIdx.x & 63) << 3) + wave) * 16;
  const int c0    = split * (N_ / NSPLIT_);
  const int lhalf = lane >> 4;
  const int l15   = lane & 15;
  const int ka    = lhalf ? 16 : 0;            // A-operand K split
  const int kb    = lhalf ? 16 : 0;            // B-operand K split
  const v8f vz    = {0.f, 0.f, 0.f, 0.f, 0.f, 0.f, 0.f, 0.f};

  float x2r[8];
#pragma unroll
  for (int v = 0; v < 8; ++v) x2r[v] = x2a[r0 + v + (lhalf << 3)];

  const size_t arow = (size_t)(r0 + l15) * D_;

  v8f o[7];                                    // 16 x 112 output accumulator
#pragma unroll
  for (int t = 0; t < 7; ++t) o[t] = vz;

  for (int ci = 0; ci < (N_ / NSPLIT_) / BN_; ++ci) {   // 16 column passes
    const int cb = c0 + ci * BN_;
    v8f a[4];
#pragma unroll
    for (int t = 0; t < 4; ++t) a[t] = vz;

    stage_chunk(xb_hi, xb_lo, &Bs[0][0][0][0], cb, 0, tid);
    for (int kc = 0; kc < D_ / KB_; ++kc) {             // 16 K chunks
      if (kc + 1 < D_ / KB_)
        stage_chunk(xb_hi, xb_lo, &Bs[(kc + 1) & 1][0][0][0], cb,
                    (kc + 1) * KB_, tid);
#if HAVE_ASYNC_LDS
      if (kc + 1 < D_ / KB_) asm volatile("s_wait_asynccnt 0x4" ::: "memory");
      else                   asm volatile("s_wait_asynccnt 0x0" ::: "memory");
#endif
      __syncthreads();                                   // chunk kc visible
      const __bf16* bsb = &Bs[kc & 1][0][0][0];
#pragma unroll
      for (int ktl = 0; ktl < KB_; ktl += 32) {
        const int kt = kc * KB_ + ktl;
        Frag Ah, Al;
        const __bf16* pah = xa_hi + arow + kt + ka;
        const __bf16* pal = xa_lo + arow + kt + ka;
        Ah.q[0] = ld16(pah); Ah.q[1] = ld16(pah + 8);
        Al.q[0] = ld16(pal); Al.q[1] = ld16(pal + 8);
#pragma unroll
        for (int t = 0; t < 4; ++t) {
          const int coll = t * 16 + l15;
          const __bf16* ph = bsb + (size_t)coll * KBP_ + ktl + kb;
          const __bf16* pl = bsb + (size_t)(BN_ + coll) * KBP_ + ktl + kb;
          Frag Bh, Bl;
          Bh.q[0] = *(const v4i*)ph; Bh.q[1] = *(const v4i*)(ph + 8);
          Bl.q[0] = *(const v4i*)pl; Bl.q[1] = *(const v4i*)(pl + 8);
          a[t] = __builtin_amdgcn_wmma_f32_16x16x32_bf16(false, Ah.bf, false, Bh.bf,
                                                         (short)0, a[t], false, false);
          a[t] = __builtin_amdgcn_wmma_f32_16x16x32_bf16(false, Ah.bf, false, Bl.bf,
                                                         (short)0, a[t], false, false);
          a[t] = __builtin_amdgcn_wmma_f32_16x16x32_bf16(false, Al.bf, false, Bh.bf,
                                                         (short)0, a[t], false, false);
        }
      }
      __syncthreads();                                   // done reading chunk kc
    }

    // Epilogue: d2 = |xi|^2 + |xj|^2 - 2G, p = exp(-sqrt(d2)), diag -> 0.
#pragma unroll
    for (int t = 0; t < 4; ++t) {
      const int   col = cb + t * 16 + l15;
      const float x2c = x2b[col];
#pragma unroll
      for (int v = 0; v < 8; ++v) {
        const int lrow = v + (lhalf << 3);
        float d2 = fmaxf(x2r[v] + x2c - 2.f * a[t][v], 0.f);
        float p  = ((r0 + lrow) == col) ? 0.f : __expf(-__builtin_sqrtf(d2));
        Pb[wave][lrow][t * 16 + l15] = (__bf16)p;
      }
    }
    asm volatile("s_wait_dscnt 0x0" ::: "memory");  // wave-local LDS RAW fence

    // GEMM2: o(16x112) += P(16x64) @ [Y|1]^T tiles (bf16 WMMA, K=64)
#pragma unroll
    for (int kt = 0; kt < BN_; kt += 32) {
      Frag Pa;
      const __bf16* pp = &Pb[wave][l15][kt + ka];
      Pa.q[0] = *(const v4i*)pp;
      Pa.q[1] = *(const v4i*)(pp + 8);
#pragma unroll
      for (int t = 0; t < 7; ++t) {
        const __bf16* py = yT + (size_t)(t * 16 + l15) * N_ + cb + kt + kb;
        Frag Yb;
        Yb.q[0] = ld16(py); Yb.q[1] = ld16(py + 8);
        o[t] = __builtin_amdgcn_wmma_f32_16x16x32_bf16(false, Pa.bf, false, Yb.bf,
                                                       (short)0, o[t], false, false);
      }
    }
  }

  // Store split-K partials (deterministic; no atomics).
  float* pout = part + ((size_t)split * N_ + r0) * CP_;
#pragma unroll
  for (int t = 0; t < 7; ++t)
#pragma unroll
    for (int v = 0; v < 8; ++v) {
      const int lrow = v + (lhalf << 3);
      pout[(size_t)lrow * CP_ + t * 16 + l15] = o[t][v];
    }
}

// ---- reduce: sum splits, normalize by ones-column row sum ------------------
__global__ void reduce_out(const float* __restrict__ part, float* __restrict__ out) {
  const int idx = blockIdx.x * blockDim.x + threadIdx.x;   // [0, N_*C_)
  const int i = idx / C_, c = idx % C_;
  float num = 0.f, den = 0.f;
#pragma unroll
  for (int s = 0; s < NSPLIT_; ++s) {
    const float* p = part + ((size_t)s * N_ + i) * CP_;
    num += p[c];
    den += p[C_];
  }
  out[idx] = num / den;
}

extern "C" void kernel_launch(void* const* d_in, const int* in_sizes, int n_in,
                              void* d_out, int out_size, void* d_ws, size_t ws_size,
                              hipStream_t stream) {
  const float* x  = (const float*)d_in[0];
  const float* xn = (const float*)d_in[1];
  const float* yn = (const float*)d_in[2];
  char* ws = (char*)d_ws;
  const size_t SZH = (size_t)N_ * D_ * sizeof(__bf16);     // 16 MB each
  __bf16* xa_hi = (__bf16*)(ws);
  __bf16* xa_lo = (__bf16*)(ws + SZH);
  __bf16* xb_hi = (__bf16*)(ws + 2 * SZH);
  __bf16* xb_lo = (__bf16*)(ws + 3 * SZH);
  float*  x2a   = (float*) (ws + 4 * SZH);
  float*  x2b   = (float*) (ws + 4 * SZH + 32768);
  __bf16* yT    = (__bf16*)(ws + 4 * SZH + 65536);
  float*  part  = (float*) (ws + 4 * SZH + 65536 + (size_t)CP_ * N_ * sizeof(__bf16));

  prep_convert<<<N_ / 8, 256, 0, stream>>>(x,  xa_hi, xa_lo, x2a);
  prep_convert<<<N_ / 8, 256, 0, stream>>>(xn, xb_hi, xb_lo, x2b);
  prep_ypadT<<<(CP_ * N_) / 256, 256, 0, stream>>>(yn, yT);
  dist_softmax_wmma<<<(N_ / 16) * NSPLIT_ / WAVES_, 256, 0, stream>>>(
      xa_hi, xa_lo, xb_hi, xb_lo, x2a, x2b, yT, part);
  reduce_out<<<(N_ * C_) / 256, 256, 0, stream>>>(part, (float*)d_out);
}
